// Informer_83451214561996
// MI455X (gfx1250) — matrix operand order repaired and checked by reading, not compiled, not run on hardware
//
#include <hip/hip_runtime.h>
#include <hip/hip_bf16.h>
#include <math.h>

// ---------------- model dims ----------------
#define DMODEL 512
#define NHEADS 8
#define DHEAD  64
#define BV     2          // batch
#define LENC0  2048
#define LDECL  2048
#define PREDL  1024
#define COUTN  7
#define EPSL   1e-5f
#define CEILDIV(a,b) (((a)+(b)-1)/(b))

typedef unsigned short ushort_t;
typedef __attribute__((ext_vector_type(16))) __bf16 v16bf;
typedef __attribute__((ext_vector_type(8)))  float  v8f;
typedef __attribute__((ext_vector_type(4)))  int    v4i;

__device__ __forceinline__ ushort_t f2bf(float f) {
  unsigned u = __float_as_uint(f);
  u += 0x7FFFu + ((u >> 16) & 1u);           // RNE to bf16
  return (ushort_t)(u >> 16);
}

// =====================================================================
// Generic batched WMMA bf16 GEMM:  C = alpha * A(MxK) x Bt(NxK)^T + bias
// Register-blocked: one 32x32 C tile per wave (2x2 WMMA 16x16x32 tiles,
// 4 v_wmma per K-step, 2 fragment loads per WMMA). wave32.
// Requires: K % 32 == 0, M % 32 == 0, N % 32 == 0  (N=64 min used).
// Batch index decomposes as (b0,b1) = (batch/bdiv, batch%bdiv).
// =====================================================================
__global__ void k_gemm_bf16(const ushort_t* __restrict__ A,
                            const ushort_t* __restrict__ Bt,
                            float* __restrict__ C,
                            const float* __restrict__ bias,
                            int M, int N, int K, int lda, int ldb, int ldc,
                            int bdiv,
                            long long sA0, long long sA1,
                            long long sB0, long long sB1,
                            long long sC0, long long sC1,
                            float alpha)
{
  const int lane = threadIdx.x & 31;
  const int wave = threadIdx.x >> 5;
  const int nTN  = N >> 5;
  const int nT   = (M >> 5) * nTN;
  const int tile = blockIdx.x * (blockDim.x >> 5) + wave;
  if (tile >= nT) return;                        // wave-uniform exit
  const int m0 = (tile / nTN) << 5;
  const int n0 = (tile % nTN) << 5;

  const int batch = blockIdx.y;
  const int b0 = batch / bdiv, b1 = batch % bdiv;
  const ushort_t* Ab = A  + b0 * sA0 + b1 * sA1;
  const ushort_t* Bb = Bt + b0 * sB0 + b1 * sB1;
  float*          Cb = C  + b0 * sC0 + b1 * sC1;

  const int half = lane >> 4;                    // 0 or 1
  const int l15  = lane & 15;
  const int kbA  = half << 3;                    // A: lanes>=16 hold K+8..
  const int kbB  = half << 3;                    // B: lanes>=16 hold K+8..
  const ushort_t* a0p = Ab + (long long)(m0 + l15) * lda + kbA;
  const ushort_t* a1p = a0p + (long long)16 * lda;
  const ushort_t* b0p = Bb + (long long)(n0 + l15) * ldb + kbB;
  const ushort_t* b1p = b0p + (long long)16 * ldb;

  v8f acc00 = {}, acc01 = {}, acc10 = {}, acc11 = {};
  for (int k0 = 0; k0 < K; k0 += 32) {
    union { v4i i[2]; v16bf v; } a0, a1, q0, q1;
    a0.i[0] = *(const v4i*)(const void*)(a0p + k0);
    a0.i[1] = *(const v4i*)(const void*)(a0p + k0 + 16);
    a1.i[0] = *(const v4i*)(const void*)(a1p + k0);
    a1.i[1] = *(const v4i*)(const void*)(a1p + k0 + 16);
    q0.i[0] = *(const v4i*)(const void*)(b0p + k0);
    q0.i[1] = *(const v4i*)(const void*)(b0p + k0 + 16);
    q1.i[0] = *(const v4i*)(const void*)(b1p + k0);
    q1.i[1] = *(const v4i*)(const void*)(b1p + k0 + 16);
    if (k0 + 128 < K) {                          // keep L2->WGP stream ahead
      __builtin_prefetch((const void*)(a0p + k0 + 128), 0, 1);
      __builtin_prefetch((const void*)(a1p + k0 + 128), 0, 1);
      __builtin_prefetch((const void*)(b0p + k0 + 128), 0, 1);
      __builtin_prefetch((const void*)(b1p + k0 + 128), 0, 1);
    }
    acc00 = __builtin_amdgcn_wmma_f32_16x16x32_bf16(false, a0.v, false, q0.v, (short)0, acc00, false, false);
    acc01 = __builtin_amdgcn_wmma_f32_16x16x32_bf16(false, a0.v, false, q1.v, (short)0, acc01, false, false);
    acc10 = __builtin_amdgcn_wmma_f32_16x16x32_bf16(false, a1.v, false, q0.v, (short)0, acc10, false, false);
    acc11 = __builtin_amdgcn_wmma_f32_16x16x32_bf16(false, a1.v, false, q1.v, (short)0, acc11, false, false);
  }
  const int mr0 = m0 + (half << 3);
  const int nc0 = n0 + l15;
#pragma unroll
  for (int r = 0; r < 8; ++r) {
    float v00 = acc00[r] * alpha, v01 = acc01[r] * alpha;
    float v10 = acc10[r] * alpha, v11 = acc11[r] * alpha;
    if (bias) {
      float e0 = bias[nc0], e1 = bias[nc0 + 16];
      v00 += e0; v01 += e1; v10 += e0; v11 += e1;
    }
    Cb[(long long)(mr0 + r) * ldc + nc0]           = v00;
    Cb[(long long)(mr0 + r) * ldc + nc0 + 16]      = v01;
    Cb[(long long)(mr0 + r + 16) * ldc + nc0]      = v10;
    Cb[(long long)(mr0 + r + 16) * ldc + nc0 + 16] = v11;
  }
}

// ---------------- elementwise / helper kernels ----------------
__global__ void k_cvt_bf16(const float* __restrict__ x, ushort_t* __restrict__ y, long long n) {
  long long i = (long long)blockIdx.x * blockDim.x + threadIdx.x;
  if (i < n) y[i] = f2bf(x[i]);
}

// Wt[n*K + k] = W[k*N + n]  (f32 -> bf16 transpose, weights K x N)
__global__ void k_transpose_cvt(const float* __restrict__ W, ushort_t* __restrict__ Wt,
                                int K, int N) {
  long long i = (long long)blockIdx.x * blockDim.x + threadIdx.x;
  if (i >= (long long)K * N) return;
  int n = (int)(i / K), k = (int)(i % K);
  Wt[(long long)n * K + k] = f2bf(W[(long long)k * N + n]);
}

// conv weight pack: Wt[o*(3C) + t*C + c] = W[(o*C+c)*3 + t]
__global__ void k_pack_convw(const float* __restrict__ W, ushort_t* __restrict__ Wt, int C) {
  long long i = (long long)blockIdx.x * blockDim.x + threadIdx.x;
  long long tot = (long long)DMODEL * 3 * C;
  if (i >= tot) return;
  int o = (int)(i / (3 * C)); int r = (int)(i % (3 * C));
  int t = r / C, c = r % C;
  Wt[(long long)o * (3 * C) + t * C + c] = f2bf(W[((long long)o * C + c) * 3 + t]);
}

// circularly padded activations -> bf16 : xp (B, L+2, 512)
__global__ void k_wrap_pad(const float* __restrict__ x, ushort_t* __restrict__ xp, int L) {
  long long i = (long long)blockIdx.x * blockDim.x + threadIdx.x;
  long long tot = (long long)BV * (L + 2) * DMODEL;
  if (i >= tot) return;
  int d = (int)(i % DMODEL);
  long long r = i / DMODEL;
  int lp = (int)(r % (L + 2));
  int b  = (int)(r / (L + 2));
  int ls = (lp == 0) ? (L - 1) : (lp == L + 1 ? 0 : lp - 1);
  xp[i] = f2bf(x[((long long)b * L + ls) * DMODEL + d]);
}

// token conv (Cin->512, k=3 circular) + positional + temporal sinusoids
__global__ void k_embed(const float* __restrict__ x, const int* __restrict__ xm,
                        const float* __restrict__ Wtok, float* __restrict__ out,
                        int Cin, int L) {
  long long i = (long long)blockIdx.x * blockDim.x + threadIdx.x;
  long long tot = (long long)BV * L * DMODEL;
  if (i >= tot) return;
  int d = (int)(i % DMODEL);
  long long r = i / DMODEL;
  int l = (int)(r % L);
  int b = (int)(r / L);
  float s = 0.f;
  for (int t = 0; t < 3; ++t) {
    int xi = l - 1 + t; if (xi < 0) xi += L; if (xi >= L) xi -= L;
    const float* xr = x + ((long long)b * L + xi) * Cin;
    for (int c = 0; c < Cin; ++c)
      s += Wtok[((long long)d * Cin + c) * 3 + t] * xr[c];
  }
  const float freq = expf(-(float)(d & ~1) * 9.210340371976184f / (float)DMODEL);
  float pe = (d & 1) ? cosf((float)l * freq) : sinf((float)l * freq);
  const int* m = xm + ((long long)b * L + l) * 4;
  int mon = m[0], day = m[1], wday = m[2], hour = m[3];
  float te = 0.f;
  {
    float ps[4] = {(float)hour, (float)wday, (float)day, (float)mon};
    for (int q = 0; q < 4; ++q)
      te += (d & 1) ? cosf(ps[q] * freq) : sinf(ps[q] * freq);
  }
  out[i] = s + pe + te;
}

__global__ void k_layernorm(const float* __restrict__ x, const float* __restrict__ g,
                            const float* __restrict__ b, float* __restrict__ y) {
  const long long row = blockIdx.x;
  const float* xr = x + row * DMODEL;
  float* yr = y + row * DMODEL;
  __shared__ float s1[256], s2[256];
  int tid = threadIdx.x;
  float a0 = xr[tid], a1 = xr[tid + 256];
  s1[tid] = a0 + a1;
  s2[tid] = a0 * a0 + a1 * a1;
  __syncthreads();
  for (int s = 128; s; s >>= 1) {
    if (tid < s) { s1[tid] += s1[tid + s]; s2[tid] += s2[tid + s]; }
    __syncthreads();
  }
  float mu = s1[0] * (1.f / DMODEL);
  float var = s2[0] * (1.f / DMODEL) - mu * mu;
  float inv = rsqrtf(var + EPSL);
  yr[tid]       = (a0 - mu) * inv * g[tid]       + b[tid];
  yr[tid + 256] = (a1 - mu) * inv * g[tid + 256] + b[tid + 256];
}

__global__ void k_add_inplace(float* __restrict__ x, const float* __restrict__ r, long long n) {
  long long i = (long long)blockIdx.x * blockDim.x + threadIdx.x;
  if (i < n) x[i] += r[i];
}

__global__ void k_gelu(float* __restrict__ x, long long n) {
  long long i = (long long)blockIdx.x * blockDim.x + threadIdx.x;
  if (i < n) { float v = x[i]; x[i] = 0.5f * v * (1.f + erff(v * 0.70710678118654752f)); }
}

__global__ void k_softmax(float* __restrict__ S, int cols) {
  float* p = S + (long long)blockIdx.x * cols;
  __shared__ float red[256];
  int tid = threadIdx.x;
  float mx = -INFINITY;
  for (int i = tid; i < cols; i += 256) mx = fmaxf(mx, p[i]);
  red[tid] = mx; __syncthreads();
  for (int s = 128; s; s >>= 1) { if (tid < s) red[tid] = fmaxf(red[tid], red[tid + s]); __syncthreads(); }
  mx = red[0]; __syncthreads();
  float sum = 0.f;
  for (int i = tid; i < cols; i += 256) { float e = expf(p[i] - mx); p[i] = e; sum += e; }
  red[tid] = sum; __syncthreads();
  for (int s = 128; s; s >>= 1) { if (tid < s) red[tid] += red[tid + s]; __syncthreads(); }
  float inv = 1.f / red[0];
  for (int i = tid; i < cols; i += 256) p[i] *= inv;
}

// ---------------- ProbSparse attention pieces ----------------
__device__ __forceinline__ unsigned hash_u(unsigned x) {
  x ^= x >> 16; x *= 0x7feb352dU; x ^= x >> 15; x *= 0x846ca68bU; x ^= x >> 16; return x;
}

// M[bh][l] = max_j(qk_j) - mean_j(qk_j) over U sampled keys. One wave per (bh,l).
__global__ void k_prob_M(const float* __restrict__ Q, const float* __restrict__ Kf,
                         float* __restrict__ M, int L, int U, unsigned seed) {
  int w = blockIdx.x * (blockDim.x >> 5) + (threadIdx.x >> 5);
  if (w >= BV * NHEADS * L) return;
  int lane = threadIdx.x & 31;
  int bh = w / L, l = w % L;
  int b = bh / NHEADS, h = bh % NHEADS;
  int d = lane * 2;
  const float* qp = Q + ((long long)(b * L + l) * DMODEL) + h * DHEAD;
  float q0 = qp[d], q1 = qp[d + 1];
  float mx = -INFINITY, sm = 0.f;
  for (int j = 0; j < U; ++j) {
    int idx = (int)(hash_u((unsigned)(l * 64 + j) * 0x9E3779B9u + seed) % (unsigned)L);
    const float* kp = Kf + ((long long)(b * L + idx) * DMODEL) + h * DHEAD;
    float s = q0 * kp[d] + q1 * kp[d + 1];
    for (int m = 16; m; m >>= 1) s += __shfl_xor(s, m, 32);
    mx = fmaxf(mx, s); sm += s;
  }
  if (lane == 0) M[(long long)bh * L + l] = mx - sm / (float)U;
}

__global__ void k_topk(const float* __restrict__ M, int* __restrict__ mtop, int L, int u) {
  __shared__ float sM[2048];
  __shared__ float rv[256];
  __shared__ int   ri[256];
  int bh = blockIdx.x, tid = threadIdx.x;
  for (int i = tid; i < 2048; i += 256) sM[i] = (i < L) ? M[(long long)bh * L + i] : -INFINITY;
  __syncthreads();
  for (int t = 0; t < u; ++t) {
    float bv = -INFINITY; int bi = 0;
    for (int i = tid; i < L; i += 256) { float v = sM[i]; if (v > bv) { bv = v; bi = i; } }
    rv[tid] = bv; ri[tid] = bi; __syncthreads();
    for (int s = 128; s; s >>= 1) {
      if (tid < s && (rv[tid + s] > rv[tid] || (rv[tid + s] == rv[tid] && ri[tid + s] < ri[tid]))) {
        rv[tid] = rv[tid + s]; ri[tid] = ri[tid + s];
      }
      __syncthreads();
    }
    if (tid == 0) { mtop[bh * u + t] = ri[0]; sM[ri[0]] = -INFINITY; }
    __syncthreads();
  }
}

__global__ void k_scores(const float* __restrict__ Q, const float* __restrict__ Kf,
                         const int* __restrict__ mtop, float* __restrict__ S,
                         int L, int u, int masked) {
  int bhu = blockIdx.x, bh = bhu / u, ui = bhu % u;
  int b = bh / NHEADS, h = bh % NHEADS;
  __shared__ float qv[DHEAD];
  int qpos = mtop[bh * u + ui];
  if (threadIdx.x < DHEAD)
    qv[threadIdx.x] = Q[((long long)(b * L + qpos) * DMODEL) + h * DHEAD + threadIdx.x];
  __syncthreads();
  float* row = S + (long long)bhu * L;
  for (int k = threadIdx.x; k < L; k += blockDim.x) {
    const float* kp = Kf + ((long long)(b * L + k) * DMODEL) + h * DHEAD;
    float d = 0.f;
    for (int t = 0; t < DHEAD; ++t) d += qv[t] * kp[t];
    d *= 0.125f;
    if (masked && k > qpos) d = -INFINITY;
    row[k] = d;
  }
}

__global__ void k_upd(const float* __restrict__ S, const float* __restrict__ V,
                      float* __restrict__ upd, int L, int u) {
  int bhu = blockIdx.x, bh = bhu / u;
  int b = bh / NHEADS, h = bh % NHEADS;
  int d = threadIdx.x;
  const float* row = S + (long long)bhu * L;
  float s = 0.f;
  for (int k = 0; k < L; ++k)
    s += row[k] * V[((long long)(b * L + k) * DMODEL) + h * DHEAD + d];
  upd[(long long)bhu * DHEAD + d] = s;
}

__global__ void k_meanv(const float* __restrict__ V, float* __restrict__ mv, int L) {
  int bh = blockIdx.x, b = bh / NHEADS, h = bh % NHEADS, d = threadIdx.x;
  float s = 0.f;
  for (int l = 0; l < L; ++l)
    s += V[((long long)(b * L + l) * DMODEL) + h * DHEAD + d];
  mv[(long long)bh * DHEAD + d] = s / (float)L;
}

__global__ void k_ctx_fill(const float* __restrict__ mv, float* __restrict__ ctx, int L) {
  long long i = (long long)blockIdx.x * blockDim.x + threadIdx.x;
  long long tot = (long long)BV * L * DMODEL;
  if (i >= tot) return;
  int hd = (int)(i % DMODEL);
  int b = (int)(i / ((long long)L * DMODEL));
  ctx[i] = mv[(long long)b * DMODEL + hd];        // mv laid out [b][h*64+d]
}

__global__ void k_ctx_cumsum(const float* __restrict__ V, float* __restrict__ ctx, int L) {
  int bh = blockIdx.x, b = bh / NHEADS, h = bh % NHEADS, d = threadIdx.x;
  float acc = 0.f;
  for (int l = 0; l < L; ++l) {
    acc += V[((long long)(b * L + l) * DMODEL) + h * DHEAD + d];
    ctx[((long long)(b * L + l) * DMODEL) + h * DHEAD + d] = acc;
  }
}

__global__ void k_scatter(float* __restrict__ ctx, const float* __restrict__ upd,
                          const int* __restrict__ mtop, int L, int u) {
  long long i = (long long)blockIdx.x * blockDim.x + threadIdx.x;
  long long tot = (long long)BV * NHEADS * u * DHEAD;
  if (i >= tot) return;
  int d = (int)(i % DHEAD);
  long long r = i / DHEAD;
  int ui = (int)(r % u);
  int bh = (int)(r / u);
  int b = bh / NHEADS, h = bh % NHEADS;
  int row = mtop[bh * u + ui];
  ctx[((long long)(b * L + row) * DMODEL) + h * DHEAD + d] = upd[((long long)bh * u + ui) * DHEAD + d];
}

// pack V (B,Lc,H*D f32) -> Vt (B,H,D,Lc bf16)
__global__ void k_pack_vt(const float* __restrict__ V, ushort_t* __restrict__ Vt, int Lc) {
  long long i = (long long)blockIdx.x * blockDim.x + threadIdx.x;
  long long tot = (long long)BV * NHEADS * DHEAD * Lc;
  if (i >= tot) return;
  int l = (int)(i % Lc);
  long long r = i / Lc;
  int d = (int)(r % DHEAD);
  long long r2 = r / DHEAD;
  int h = (int)(r2 % NHEADS);
  int b = (int)(r2 / NHEADS);
  Vt[i] = f2bf(V[((long long)(b * Lc + l) * DMODEL) + h * DHEAD + d]);
}

// ---------------- distil (BN over (B,L), ELU, maxpool) ----------------
__global__ void k_bn_stats(const float* __restrict__ y, float* __restrict__ mu,
                           float* __restrict__ var, int rows) {
  int c = blockIdx.x, tid = threadIdx.x;
  __shared__ float s1[256], s2[256];
  float a = 0.f, b = 0.f;
  for (int r = tid; r < rows; r += 256) {
    float v = y[(long long)r * DMODEL + c];
    a += v; b += v * v;
  }
  s1[tid] = a; s2[tid] = b; __syncthreads();
  for (int s = 128; s; s >>= 1) { if (tid < s) { s1[tid] += s1[tid + s]; s2[tid] += s2[tid + s]; } __syncthreads(); }
  if (tid == 0) {
    float m = s1[0] / (float)rows;
    mu[c] = m;
    var[c] = s2[0] / (float)rows - m * m;
  }
}

__global__ void k_bn_elu(float* __restrict__ y, const float* __restrict__ mu,
                         const float* __restrict__ var, const float* __restrict__ g,
                         const float* __restrict__ be, long long n) {
  long long i = (long long)blockIdx.x * blockDim.x + threadIdx.x;
  if (i >= n) return;
  int c = (int)(i % DMODEL);
  float v = (y[i] - mu[c]) * rsqrtf(var[c] + EPSL) * g[c] + be[c];
  y[i] = v > 0.f ? v : expf(v) - 1.f;
}

__global__ void k_maxpool(const float* __restrict__ y, float* __restrict__ out, int L) {
  int Lo = L / 2;
  long long i = (long long)blockIdx.x * blockDim.x + threadIdx.x;
  long long tot = (long long)BV * Lo * DMODEL;
  if (i >= tot) return;
  int c = (int)(i % DMODEL);
  long long r = i / DMODEL;
  int lp = (int)(r % Lo);
  int b = (int)(r / Lo);
  float m = -INFINITY;
  for (int t = -1; t <= 1; ++t) {
    int l = 2 * lp + t;
    if (l >= 0 && l < L) m = fmaxf(m, y[((long long)b * L + l) * DMODEL + c]);
  }
  out[i] = m;
}

// final projection 512 -> 7, keep last PREDL rows
__global__ void k_proj_out(const float* __restrict__ x, const float* __restrict__ Wp,
                           const float* __restrict__ bp, float* __restrict__ out) {
  long long i = (long long)blockIdx.x * blockDim.x + threadIdx.x;
  long long tot = (long long)BV * PREDL * COUTN;
  if (i >= tot) return;
  int c = (int)(i % COUTN);
  long long r = i / COUTN;
  int l = (int)(r % PREDL);
  int b = (int)(r / PREDL);
  const float* xr = x + ((long long)b * LDECL + (LDECL - PREDL + l)) * DMODEL;
  float s = bp[c];
  for (int d = 0; d < DMODEL; ++d) s += xr[d] * Wp[(long long)d * COUTN + c];
  out[i] = s;
}

// =====================================================================
// Host side
// =====================================================================
struct AttnP { const float *Wq,*bq,*Wk,*bk,*Wv,*bv,*Wo,*bo; };
struct Ctx {
  hipStream_t s;
  float *X,*R,*Q,*K,*V,*CTX,*T1,*ENC,*SCORES;
  ushort_t *BF0,*BF1,*BF2,*ATTNBF;
  int* mtop; float *Marr,*upd,*meanV,*bmu,*bvar;
};

static void cvt(const Ctx& c, const float* x, ushort_t* y, long long n) {
  k_cvt_bf16<<<(unsigned)CEILDIV(n, 256), 256, 0, c.s>>>(x, y, n);
}
static void gemm(const Ctx& c, const ushort_t* A, const ushort_t* Bt, float* C,
                 const float* bias, int M, int N, int K, int lda, int ldb, int ldc,
                 int batch, int bdiv,
                 long long sA0, long long sA1, long long sB0, long long sB1,
                 long long sC0, long long sC1, float alpha) {
  int tiles = (M / 32) * (N / 32);
  dim3 g(CEILDIV(tiles, 8), batch);
  k_gemm_bf16<<<g, 256, 0, c.s>>>(A, Bt, C, bias, M, N, K, lda, ldb, ldc,
                                  bdiv, sA0, sA1, sB0, sB1, sC0, sC1, alpha);
}
// Out(MxN) = X(Mx512) @ W(512xN) + b
static void proj512(const Ctx& c, const float* X, const float* W, const float* b,
                    float* Out, int M, int N) {
  cvt(c, X, c.BF0, (long long)M * DMODEL);
  k_transpose_cvt<<<(unsigned)CEILDIV((long long)DMODEL * N, 256), 256, 0, c.s>>>(W, c.BF1, DMODEL, N);
  gemm(c, c.BF0, c.BF1, Out, b, M, N, DMODEL, DMODEL, DMODEL, N,
       1, 1, 0, 0, 0, 0, 0, 0, 1.f);
}
static void layernorm(const Ctx& c, const float* x, const float* g, const float* b,
                      float* y, int rows) {
  k_layernorm<<<rows, 256, 0, c.s>>>(x, g, b, y);
}
static void prob_attn(const Ctx& c, int L, int u, int masked, unsigned seed) {
  const int BH = BV * NHEADS;
  k_prob_M<<<(unsigned)CEILDIV(BH * L, 8), 256, 0, c.s>>>(c.Q, c.K, c.Marr, L, u, seed);
  k_topk<<<BH, 256, 0, c.s>>>(c.Marr, c.mtop, L, u);
  k_scores<<<BH * u, 128, 0, c.s>>>(c.Q, c.K, c.mtop, c.SCORES, L, u, masked);
  k_softmax<<<BH * u, 256, 0, c.s>>>(c.SCORES, L);
  k_upd<<<BH * u, DHEAD, 0, c.s>>>(c.SCORES, c.V, c.upd, L, u);
  if (masked) {
    k_ctx_cumsum<<<BH, DHEAD, 0, c.s>>>(c.V, c.CTX, L);
  } else {
    k_meanv<<<BH, DHEAD, 0, c.s>>>(c.V, c.meanV, L);
    k_ctx_fill<<<(unsigned)CEILDIV((long long)BV * L * DMODEL, 256), 256, 0, c.s>>>(c.meanV, c.CTX, L);
  }
  k_scatter<<<(unsigned)CEILDIV(BH * u * DHEAD, 256), 256, 0, c.s>>>(c.CTX, c.upd, c.mtop, L, u);
}
// full cross-attention: Q (B,LDECL,512), K/V (B,Lc,512) -> CTX (B,LDECL,512)
static void cross_attn(const Ctx& c, int Lc) {
  const long long Lq = LDECL;
  cvt(c, c.Q, c.BF0, (long long)BV * Lq * DMODEL);
  cvt(c, c.K, c.BF1, (long long)BV * Lc * DMODEL);
  k_pack_vt<<<(unsigned)CEILDIV((long long)BV * NHEADS * DHEAD * Lc, 256), 256, 0, c.s>>>(c.V, c.BF2, Lc);
  // scores = Q K^T / 8
  gemm(c, c.BF0, c.BF1, c.SCORES, nullptr, (int)Lq, Lc, DHEAD,
       DMODEL, DMODEL, Lc, BV * NHEADS, NHEADS,
       Lq * DMODEL, DHEAD, (long long)Lc * DMODEL, DHEAD,
       (long long)NHEADS * Lq * Lc, Lq * Lc, 0.125f);
  k_softmax<<<(unsigned)(BV * NHEADS * Lq), 256, 0, c.s>>>(c.SCORES, Lc);
  cvt(c, c.SCORES, c.ATTNBF, (long long)BV * NHEADS * Lq * Lc);
  // CTX = attn @ V   (written straight back into (B,L,H*D) layout)
  gemm(c, c.ATTNBF, c.BF2, c.CTX, nullptr, (int)Lq, DHEAD, Lc,
       Lc, Lc, DMODEL, BV * NHEADS, NHEADS,
       (long long)NHEADS * Lq * Lc, Lq * Lc,
       (long long)NHEADS * DHEAD * Lc, (long long)DHEAD * Lc,
       Lq * DMODEL, DHEAD, 1.f);
}
static void ffn_block(const Ctx& c, const float* W1, const float* b1,
                      const float* W2, const float* b2,
                      const float* g, const float* be, int M) {
  proj512(c, c.X, W1, b1, c.T1, M, DMODEL);
  k_gelu<<<(unsigned)CEILDIV((long long)M * DMODEL, 256), 256, 0, c.s>>>(c.T1, (long long)M * DMODEL);
  proj512(c, c.T1, W2, b2, c.R, M, DMODEL);
  k_add_inplace<<<(unsigned)CEILDIV((long long)M * DMODEL, 256), 256, 0, c.s>>>(c.X, c.R, (long long)M * DMODEL);
  layernorm(c, c.X, g, be, c.X, M);
}

extern "C" void kernel_launch(void* const* d_in, const int* in_sizes, int n_in,
                              void* d_out, int out_size, void* d_ws, size_t ws_size,
                              hipStream_t stream) {
  if (n_in < 120) return;
  const size_t F = (size_t)BV * 2048 * DMODEL * sizeof(float);  // 8 MiB
  if (ws_size < 23 * F) return;

  // ---- inputs ----
  const float* x_enc = (const float*)d_in[0];
  const int*   xm_enc = (const int*)d_in[1];
  const float* x_dec = (const float*)d_in[2];
  const int*   xm_dec = (const int*)d_in[3];
  int pi = 4;
  auto nf = [&]() { return (const float*)d_in[pi++]; };
  const float* enc_tok_W = nf();
  const float* dec_tok_W = nf();
  struct { const float *W1,*b1,*W2,*b2,*g1,*be1,*g2,*be2; AttnP attn; } enc[3];
  for (int i = 0; i < 3; ++i) {
    enc[i].W1 = nf(); enc[i].b1 = nf(); enc[i].W2 = nf(); enc[i].b2 = nf();
    enc[i].g1 = nf(); enc[i].be1 = nf(); enc[i].g2 = nf(); enc[i].be2 = nf();
    AttnP& a = enc[i].attn;
    a.Wq = nf(); a.bq = nf(); a.Wk = nf(); a.bk = nf();
    a.Wv = nf(); a.bv = nf(); a.Wo = nf(); a.bo = nf();
  }
  struct { const float *W,*b,*gamma,*beta; } conv[2];
  for (int i = 0; i < 2; ++i) { conv[i].W = nf(); conv[i].b = nf(); conv[i].gamma = nf(); conv[i].beta = nf(); }
  const float* enc_ng = nf(); const float* enc_nb = nf();
  struct { const float *W1,*b1,*W2,*b2,*g1,*be1,*g2,*be2,*g3,*be3; AttnP self_, cross; } dec[2];
  for (int i = 0; i < 2; ++i) {
    dec[i].W1 = nf(); dec[i].b1 = nf(); dec[i].W2 = nf(); dec[i].b2 = nf();
    dec[i].g1 = nf(); dec[i].be1 = nf(); dec[i].g2 = nf(); dec[i].be2 = nf();
    dec[i].g3 = nf(); dec[i].be3 = nf();
    AttnP* as[2] = { &dec[i].self_, &dec[i].cross };
    for (int j = 0; j < 2; ++j) {
      as[j]->Wq = nf(); as[j]->bq = nf(); as[j]->Wk = nf(); as[j]->bk = nf();
      as[j]->Wv = nf(); as[j]->bv = nf(); as[j]->Wo = nf(); as[j]->bo = nf();
    }
  }
  const float* dec_ng = nf(); const float* dec_nb = nf();
  const float* Wp = nf(); const float* bp = nf();

  // ---- workspace carve ----
  char* ws = (char*)d_ws;
  Ctx c;
  c.s   = stream;
  c.X   = (float*)(ws + 0 * F);
  c.R   = (float*)(ws + 1 * F);
  c.Q   = (float*)(ws + 2 * F);
  c.K   = (float*)(ws + 3 * F);
  c.V   = (float*)(ws + 4 * F);
  c.CTX = (float*)(ws + 5 * F);
  c.T1  = (float*)(ws + 6 * F);
  c.ENC = (float*)(ws + 7 * F);
  c.BF0 = (ushort_t*)(ws + 8 * F);            // 8 MiB (fits wrap-padded im2col)
  c.BF1 = (ushort_t*)(ws + 9 * F);            // 4 MiB
  c.BF2 = (ushort_t*)(ws + 9 * F + F / 2);    // 4 MiB
  c.ATTNBF = (ushort_t*)(ws + 10 * F);        // 32 MiB
  c.SCORES = (float*)(ws + 14 * F);           // 64 MiB
  char* MISC = ws + 22 * F;                   // 8 MiB
  c.mtop  = (int*)(MISC);
  c.Marr  = (float*)(MISC + 64 * 1024);
  c.upd   = (float*)(MISC + 256 * 1024);
  c.meanV = (float*)(MISC + 512 * 1024);
  c.bmu   = (float*)(MISC + 576 * 1024);
  c.bvar  = (float*)(MISC + 640 * 1024);

  // ================= encoder =================
  int L = LENC0;
  k_embed<<<(unsigned)CEILDIV((long long)BV * L * DMODEL, 256), 256, 0, stream>>>(
      x_enc, xm_enc, enc_tok_W, c.X, 7, L);
  unsigned cseed = 0;
  for (int i = 0; i < 3; ++i) {
    int M = BV * L;
    int u = 5 * (int)ceil(log((double)L)); if (u > L) u = L;
    proj512(c, c.X, enc[i].attn.Wq, enc[i].attn.bq, c.Q, M, DMODEL);
    proj512(c, c.X, enc[i].attn.Wk, enc[i].attn.bk, c.K, M, DMODEL);
    proj512(c, c.X, enc[i].attn.Wv, enc[i].attn.bv, c.V, M, DMODEL);
    prob_attn(c, L, u, 0, 0x9E3779B9u * (cseed + 1));
    cseed++;
    proj512(c, c.CTX, enc[i].attn.Wo, enc[i].attn.bo, c.R, M, DMODEL);
    k_add_inplace<<<(unsigned)CEILDIV((long long)M * DMODEL, 256), 256, 0, stream>>>(c.X, c.R, (long long)M * DMODEL);
    layernorm(c, c.X, enc[i].g1, enc[i].be1, c.X, M);
    ffn_block(c, enc[i].W1, enc[i].b1, enc[i].W2, enc[i].b2, enc[i].g2, enc[i].be2, M);
    if (i < 2) { // distil: circular conv (im2col GEMM) + BN + ELU + maxpool/2
      k_wrap_pad<<<(unsigned)CEILDIV((long long)BV * (L + 2) * DMODEL, 256), 256, 0, stream>>>(c.X, c.BF0, L);
      k_pack_convw<<<(unsigned)CEILDIV((long long)DMODEL * 3 * DMODEL, 256), 256, 0, stream>>>(conv[i].W, c.BF1, DMODEL);
      gemm(c, c.BF0, c.BF1, c.T1, conv[i].b, L, DMODEL, 3 * DMODEL,
           DMODEL, 3 * DMODEL, DMODEL, BV, 1,
           (long long)(L + 2) * DMODEL, 0, 0, 0, (long long)L * DMODEL, 0, 1.f);
      k_bn_stats<<<DMODEL, 256, 0, stream>>>(c.T1, c.bmu, c.bvar, BV * L);
      k_bn_elu<<<(unsigned)CEILDIV((long long)BV * L * DMODEL, 256), 256, 0, stream>>>(
          c.T1, c.bmu, c.bvar, conv[i].gamma, conv[i].beta, (long long)BV * L * DMODEL);
      k_maxpool<<<(unsigned)CEILDIV((long long)BV * (L / 2) * DMODEL, 256), 256, 0, stream>>>(c.T1, c.X, L);
      L /= 2;
    }
  }
  layernorm(c, c.X, enc_ng, enc_nb, c.ENC, BV * L);   // enc output, Lc = L = 512
  const int Lc = L;

  // ================= decoder =================
  const int Ld = LDECL, Md = BV * Ld;
  k_embed<<<(unsigned)CEILDIV((long long)BV * Ld * DMODEL, 256), 256, 0, stream>>>(
      x_dec, xm_dec, dec_tok_W, c.X, 7, Ld);
  for (int i = 0; i < 2; ++i) {
    int u = 5 * (int)ceil(log((double)Ld)); if (u > Ld) u = Ld;
    // masked self prob-attention
    proj512(c, c.X, dec[i].self_.Wq, dec[i].self_.bq, c.Q, Md, DMODEL);
    proj512(c, c.X, dec[i].self_.Wk, dec[i].self_.bk, c.K, Md, DMODEL);
    proj512(c, c.X, dec[i].self_.Wv, dec[i].self_.bv, c.V, Md, DMODEL);
    prob_attn(c, Ld, u, 1, 0x9E3779B9u * (cseed + 1));
    cseed++;
    proj512(c, c.CTX, dec[i].self_.Wo, dec[i].self_.bo, c.R, Md, DMODEL);
    k_add_inplace<<<(unsigned)CEILDIV((long long)Md * DMODEL, 256), 256, 0, stream>>>(c.X, c.R, (long long)Md * DMODEL);
    layernorm(c, c.X, dec[i].g1, dec[i].be1, c.X, Md);
    // full cross-attention vs encoder output
    proj512(c, c.X,  dec[i].cross.Wq, dec[i].cross.bq, c.Q, Md, DMODEL);
    proj512(c, c.ENC, dec[i].cross.Wk, dec[i].cross.bk, c.K, BV * Lc, DMODEL);
    proj512(c, c.ENC, dec[i].cross.Wv, dec[i].cross.bv, c.V, BV * Lc, DMODEL);
    cross_attn(c, Lc);
    proj512(c, c.CTX, dec[i].cross.Wo, dec[i].cross.bo, c.R, Md, DMODEL);
    k_add_inplace<<<(unsigned)CEILDIV((long long)Md * DMODEL, 256), 256, 0, stream>>>(c.X, c.R, (long long)Md * DMODEL);
    layernorm(c, c.X, dec[i].g2, dec[i].be2, c.X, Md);
    ffn_block(c, dec[i].W1, dec[i].b1, dec[i].W2, dec[i].b2, dec[i].g3, dec[i].be3, Md);
  }
  layernorm(c, c.X, dec_ng, dec_nb, c.X, Md);
  k_proj_out<<<(unsigned)CEILDIV((long long)BV * PREDL * COUTN, 256), 256, 0, stream>>>(
      c.X, Wp, bp, (float*)d_out);
  (void)in_sizes; (void)out_size;
}
